// BaseModelWithLSA_34230889349532
// MI455X (gfx1250) — matrix operand order repaired and checked by reading, not compiled, not run on hardware
//
#include <hip/hip_runtime.h>
#include <hip/hip_bf16.h>

typedef __attribute__((ext_vector_type(16))) _Float16 v16h;
typedef __attribute__((ext_vector_type(8)))  _Float16 v8h;
typedef __attribute__((ext_vector_type(4)))  _Float16 v4h;
typedef __attribute__((ext_vector_type(8)))  float    v8f;
typedef __attribute__((ext_vector_type(4)))  float    v4f;

#define BATCH 64
#define SEQ   1024
#define DIM   768
#define QTILE 16          // queries per workgroup
#define KEYS_PER_WAVE 128 // phase-1 key span per wave (8 tiles of 16)
#define D_PER_WAVE    96  // phase-2 output-dim span per wave (6 tiles of 16)
#define NDC   (DIM / 32)  // 24 K-chunks in phase 1
#define NKC   (SEQ / 32)  // 32 key-chunks in phase 2

// ---------------------------------------------------------------------------
// Preprocess 1: elementwise f32 -> f16 copy of K (row-major [b][key][d])
// ---------------------------------------------------------------------------
__global__ __launch_bounds__(256)
void cvt_f32_to_f16(const float* __restrict__ src, _Float16* __restrict__ dst,
                    int n4) {
    int i = blockIdx.x * blockDim.x + threadIdx.x;
    if (i >= n4) return;
    v4f a = ((const v4f*)src)[i];
    v4h r;
#pragma unroll
    for (int j = 0; j < 4; ++j) r[j] = (_Float16)a[j];
    ((v4h*)dst)[i] = r;
}

// ---------------------------------------------------------------------------
// Preprocess 2: V[b][k][d] f32 -> Vt[b][d][k] f16 (LDS-tiled 32x32 transpose)
// ---------------------------------------------------------------------------
__global__ __launch_bounds__(256)
void transpose_v_f16(const float* __restrict__ V, _Float16* __restrict__ Vt) {
    __shared__ _Float16 tile[32][33];  // +1 pad: no LDS bank conflicts
    const int blk = blockIdx.x;
    const int b   = blk / ((SEQ / 32) * (DIM / 32));
    const int rem = blk % ((SEQ / 32) * (DIM / 32));
    const int kt  = rem / (DIM / 32);
    const int dt  = rem % (DIM / 32);

    const float* Vb  = V  + (size_t)b * SEQ * DIM;
    _Float16*    Vtb = Vt + (size_t)b * DIM * SEQ;

    const int tx = threadIdx.x & 31;
    const int ty = threadIdx.x >> 5;  // 0..7
    const int k0 = kt * 32, d0 = dt * 32;

#pragma unroll
    for (int i = 0; i < 4; ++i) {
        const int k = ty + 8 * i;
        tile[k][tx] = (_Float16)Vb[(size_t)(k0 + k) * DIM + d0 + tx];
    }
    __syncthreads();
#pragma unroll
    for (int i = 0; i < 4; ++i) {
        const int d = ty + 8 * i;
        Vtb[(size_t)(d0 + d) * SEQ + k0 + tx] = tile[tx][d];
    }
}

// ---------------------------------------------------------------------------
// Helpers: build v16h fragments
// ---------------------------------------------------------------------------
// 16 consecutive f32 (p..p+15) -> v16h  (contiguous k order: lane half
// h holds k = {h*16..h*16+15} of the 32-wide K chunk)
__device__ __forceinline__ v16h cvt8x2(const float* __restrict__ p) {
    const v4f* q0 = (const v4f*)p;
    const v4f* q1 = (const v4f*)(p + 8);
    v4f a0 = q0[0], a1 = q0[1], a2 = q1[0], a3 = q1[1];
    v16h r;
#pragma unroll
    for (int j = 0; j < 4; ++j) {
        r[j]      = (_Float16)a0[j];
        r[4 + j]  = (_Float16)a1[j];
        r[8 + j]  = (_Float16)a2[j];
        r[12 + j] = (_Float16)a3[j];
    }
    return r;
}

// 16 consecutive f16 -> v16h (two 16B loads, no conversion)
__device__ __forceinline__ v16h ld16h(const _Float16* __restrict__ p) {
    v8h a = *(const v8h*)p;
    v8h b = *(const v8h*)(p + 8);
    v16h r;
#pragma unroll
    for (int j = 0; j < 8; ++j) { r[j] = a[j]; r[8 + j] = b[j]; }
    return r;
}

// ---------------------------------------------------------------------------
// Fused flash attention with diagonal self-exclusion
// ---------------------------------------------------------------------------
__global__ __launch_bounds__(256, 1)
void lsa_flash_attn(const float* __restrict__ Q, const _Float16* __restrict__ Kh,
                    const _Float16* __restrict__ Vt, const float* __restrict__ temp,
                    float* __restrict__ Out) {
    __shared__ _Float16 Pbuf[QTILE * SEQ];  // scores -> exp weights, f16, 32KB
    __shared__ float    rowinv[QTILE];      // per-row 1/sum(exp)

    const int b    = blockIdx.x / (SEQ / QTILE);
    const int qt   = blockIdx.x % (SEQ / QTILE);
    const int tid  = threadIdx.x;
    const int wave = tid >> 5;
    const int lane = tid & 31;
    const int lh   = lane >> 4;   // lane half (0/1)
    const int l15  = lane & 15;

    const float invT = 1.0f / temp[0];

    const float*    Qb  = Q  + ((size_t)b * SEQ + (size_t)qt * QTILE) * DIM;
    const _Float16* Khb = Kh + (size_t)b * SEQ * DIM;   // [key][d] f16
    const _Float16* Vtb = Vt + (size_t)b * DIM * SEQ;   // [d][key] f16

    // ---------------- Phase 1: S[16 x 1024] = Q Kt * invT ----------------
    // wave owns keys [wave*128, wave*128+128): 8 accumulator tiles of 16x16.
    // B fragments use a two-slot alternating buffer (slot = nt&1): after full
    // unroll the slot index is constant, so prefetch loads write directly
    // into the registers WMMA reads two iterations later — no ring copies.
    {
        v8f acc[8];
#pragma unroll
        for (int i = 0; i < 8; ++i) acc[i] = (v8f){0, 0, 0, 0, 0, 0, 0, 0};

        const int key0 = wave * KEYS_PER_WAVE;
        const _Float16* kb = Khb + (size_t)(key0 + l15) * DIM + lh * 16;
        const float*    qb = Qb  + (size_t)l15 * DIM + lh * 16;

        v16h af = cvt8x2(qb);                            // (dc=0)
        v16h bfr[2];
        bfr[0] = ld16h(kb);                              // (dc=0, nt=0)
        bfr[1] = ld16h(kb + (size_t)16 * DIM);           // (dc=0, nt=1)

        for (int dc = 0; dc < NDC; ++dc) {
            v16h afn = af;
            if (dc + 1 < NDC) afn = cvt8x2(qb + (dc + 1) * 32);
#pragma unroll
            for (int nt = 0; nt < 8; ++nt) {
                acc[nt] = __builtin_amdgcn_wmma_f32_16x16x32_f16(
                    false, af, false, bfr[nt & 1], (short)0, acc[nt],
                    false, false);
                // refill the slot just consumed with fragment (nt+2)
                const int linear = nt + 2;
                const int pnt = (linear < 8) ? linear : linear - 8;
                const int pdc = (linear < 8) ? dc
                              : ((dc + 1 < NDC) ? dc + 1 : dc);
                bfr[nt & 1] = ld16h(kb + (size_t)pnt * 16 * DIM + pdc * 32);
            }
            af = afn;
        }
        // C layout: VGPR r, lane half h -> row m = r + 8h, col n = l15
#pragma unroll
        for (int nt = 0; nt < 8; ++nt) {
            const int col = key0 + nt * 16 + l15;
#pragma unroll
            for (int r = 0; r < 8; ++r) {
                const int m = r + 8 * lh;
                Pbuf[m * SEQ + col] = (_Float16)(acc[nt][r] * invT);
            }
        }
    }
    __syncthreads();

    // ---------------- Softmax (unnormalized exp; keep 1/sum) -------------
    // wave owns rows {2*wave, 2*wave+1}
#pragma unroll
    for (int rr = 0; rr < 2; ++rr) {
        const int m     = wave * 2 + rr;
        const int qglob = qt * QTILE + m;
        if (lane == 0) Pbuf[m * SEQ + qglob] = (_Float16)(-__builtin_inff());
        // DS ops are in-order within a wave; loads below see the store above.
        float mx = -__builtin_inff();
        for (int c = lane; c < SEQ; c += 32)
            mx = fmaxf(mx, (float)Pbuf[m * SEQ + c]);
#pragma unroll
        for (int off = 16; off > 0; off >>= 1)
            mx = fmaxf(mx, __shfl_xor(mx, off, 32));
        float sum = 0.0f;
        for (int c = lane; c < SEQ; c += 32) {
            float e = __expf((float)Pbuf[m * SEQ + c] - mx);
            Pbuf[m * SEQ + c] = (_Float16)e;
            sum += e;
        }
#pragma unroll
        for (int off = 16; off > 0; off >>= 1)
            sum += __shfl_xor(sum, off, 32);
        if (lane == 0) rowinv[m] = 1.0f / sum;
    }
    __syncthreads();

    // ---------------- Phase 2: Out[16 x 768] = P V  ----------------------
    // wave owns output dims [wave*96, wave*96+96): 6 accumulator tiles.
    // Same alternating two-slot pipeline for the Vt B fragments.
    {
        float rinv[8];
#pragma unroll
        for (int r = 0; r < 8; ++r) rinv[r] = rowinv[r + 8 * lh];

        v8f oacc[6];
#pragma unroll
        for (int i = 0; i < 6; ++i) oacc[i] = (v8f){0, 0, 0, 0, 0, 0, 0, 0};

        const int d0 = wave * D_PER_WAVE;
        const _Float16* vb = Vtb + (size_t)(d0 + l15) * SEQ + lh * 16;

        v16h bfr[2];
        bfr[0] = ld16h(vb);                              // (kc=0, nt=0)
        bfr[1] = ld16h(vb + (size_t)16 * SEQ);           // (kc=0, nt=1)

        for (int kc = 0; kc < NKC; ++kc) {
            // A fragment from LDS: P row m = l15, contiguous k order (k = 16h + s) over key chunk
            const _Float16* pp = &Pbuf[l15 * SEQ + kc * 32 + lh * 16];
            v8h p0 = *(const v8h*)pp;
            v8h p1 = *(const v8h*)(pp + 8);
            v16h af;
#pragma unroll
            for (int j = 0; j < 8; ++j) { af[j] = p0[j]; af[8 + j] = p1[j]; }

#pragma unroll
            for (int nt = 0; nt < 6; ++nt) {
                oacc[nt] = __builtin_amdgcn_wmma_f32_16x16x32_f16(
                    false, af, false, bfr[nt & 1], (short)0, oacc[nt],
                    false, false);
                const int linear = nt + 2;
                const int pnt = (linear < 6) ? linear : linear - 6;
                const int pkc = (linear < 6) ? kc
                              : ((kc + 1 < NKC) ? kc + 1 : kc);
                bfr[nt & 1] = ld16h(vb + (size_t)pnt * 16 * SEQ + pkc * 32);
            }
        }

        float* op = Out + ((size_t)b * SEQ + (size_t)qt * QTILE) * DIM + d0;
#pragma unroll
        for (int nt = 0; nt < 6; ++nt) {
#pragma unroll
            for (int r = 0; r < 8; ++r) {
                const int m = r + 8 * lh;
                op[(size_t)m * DIM + nt * 16 + l15] = oacc[nt][r] * rinv[r];
            }
        }
    }
}

extern "C" void kernel_launch(void* const* d_in, const int* in_sizes, int n_in,
                              void* d_out, int out_size, void* d_ws, size_t ws_size,
                              hipStream_t stream) {
    const float* Q    = (const float*)d_in[0];
    const float* K    = (const float*)d_in[1];
    const float* V    = (const float*)d_in[2];
    const float* temp = (const float*)d_in[3];
    float* out        = (float*)d_out;

    // Workspace: Kh f16 (100.7 MB) | Vt f16 (100.7 MB)
    const size_t elems = (size_t)BATCH * SEQ * DIM;
    _Float16* Kh = (_Float16*)d_ws;
    _Float16* Vt = Kh + elems;

    // K f32 -> f16 (row-major), exact multiple: elems/4 threads
    {
        const int n4 = (int)(elems / 4);
        cvt_f32_to_f16<<<dim3(n4 / 256), dim3(256), 0, stream>>>(K, Kh, n4);
    }
    // V f32 [b][k][d] -> f16 transposed [b][d][k]
    {
        const int nblk = BATCH * (SEQ / 32) * (DIM / 32);
        transpose_v_f16<<<dim3(nblk), dim3(256), 0, stream>>>(V, Vt);
    }
    // Fused attention
    lsa_flash_attn<<<dim3(BATCH * (SEQ / QTILE)), dim3(256), 0, stream>>>(
        Q, Kh, Vt, temp, out);
}
